// TCN_MAE_19533511262662
// MI455X (gfx1250) — matrix operand order repaired and checked by reading, not compiled, not run on hardware
//
#include <hip/hip_runtime.h>
#include <hip/hip_bf16.h>

typedef __attribute__((ext_vector_type(16))) _Float16 v16h;
typedef __attribute__((ext_vector_type(8)))  float    v8f;

// ---------------- problem constants ----------------
constexpr int  Lx   = 1024;
constexpr int  Bx   = 64;
constexpr int  PADx = 64;            // >= 2*max_dil = 64
constexpr int  PLx  = PADx + Lx + PADx;   // 1152 padded row length (f16 elems)
constexpr long SB   = 32L * PLx;     // one 32-channel sample block

// workspace layout (units: f16 halves)
constexpr long OFF_CBUF  = 0;                                  // [B][128][PL]
constexpr long OFF_FEATE = OFF_CBUF  + (long)Bx * 128 * PLx;   // [6][B][32][PL]
constexpr long OFF_ENC   = OFF_FEATE + 6L * Bx * 32 * PLx;     // [B][64][PL]
constexpr long OFF_HDEC  = OFF_ENC   + (long)Bx * 64 * PLx;    // [B][64][PL]
constexpr long OFF_FF    = OFF_HDEC  + (long)Bx * 64 * PLx;    // [4][B][32][PL]
constexpr long OFF_FV    = OFF_FF    + 4L * Bx * 32 * PLx;     // [8][B][32][PL] (4 combos x 2)
constexpr long OFF_WP    = OFF_FV    + 8L * Bx * 32 * PLx;     // packed weight fragments
constexpr long ZERO_HALVES = OFF_WP;                           // zero everything before WP

// packed-weight fragment bases (units: fragments of 512 halves)
constexpr long FRAG      = 512;
constexpr long FR_ENCW1  = 0;    // 5 layers x 40
constexpr long FR_ENC0W2 = 200;  // 8
constexpr long FR_ENCW2  = 208;  // 5 x 8
constexpr long FR_COMP   = 248;  // 24
constexpr long FR_DF0W1  = 272;  // 80
constexpr long FR_DF0W2  = 352;  // 8
constexpr long FR_DFW1   = 360;  // 3 x 40
constexpr long FR_DFW2   = 480;  // 3 x 8
constexpr long FR_DVW1   = 504;  // 8 x 40
constexpr long FR_DVW2   = 824;  // 8 x 8   (total 888 frags)

// ---------------- helpers ----------------
__device__ __forceinline__ v16h loadu16h(const _Float16* p) {
    v16h r;
    __builtin_memcpy(&r, p, sizeof(r));   // may be 2B-aligned (odd dil=1 shifts)
    return r;
}

// ---------------- zero workspace ----------------
__global__ void k_zero(uint4* __restrict__ p, long n) {
    long i  = (long)blockIdx.x * blockDim.x + threadIdx.x;
    long st = (long)gridDim.x * blockDim.x;
    uint4 z; z.x = 0u; z.y = 0u; z.z = 0u; z.w = 0u;
    for (; i < n; i += st) p[i] = z;
}

// ---------------- weight packer: f32 [layer][Cout][Cin][taps] -> WMMA A fragments ----------------
// A-fragment (16x32 f16): lane = {M=lane&15, hi=lane>>4}; element e -> K per ISA layout.
__global__ void k_pack(const float* __restrict__ src, _Float16* __restrict__ dst,
                       int layers, int Cout, int Cin, int taps) {
    const int lane = threadIdx.x & 31;
    const int wv   = ((int)blockIdx.x * (int)blockDim.x + (int)threadIdx.x) >> 5;
    const int mt = Cout >> 4, ch = Cin >> 5;
    const int fpl = mt * ch * taps;
    if (wv >= layers * fpl) return;
    const int layer = wv / fpl;
    int rem = wv % fpl;
    const int m = rem / (ch * taps); rem %= (ch * taps);
    const int c = rem / taps;
    const int t = rem % taps;
    const int Mr = lane & 15, hi = lane >> 4;
    v16h out;
    #pragma unroll
    for (int e = 0; e < 16; ++e) {
        const int v = e >> 1, lo = e & 1;
        const int Kc = ((v < 4) ? (2 * v + lo) : (16 + 2 * (v - 4) + lo)) + hi * 8;
        const float w = src[(((size_t)layer * Cout + m * 16 + Mr) * Cin + c * 32 + Kc) * taps + t];
        out[e] = (_Float16)w;
    }
    *(v16h*)(dst + (size_t)wv * FRAG + (size_t)lane * 16) = out;
}

// ---------------- generic WMMA dilated conv + bias + optional ReLU ----------------
// in : f16 padded activations; chunks of 32 channels at stride inChunk, samples at inSamp
// out: f16 padded [B][MT*16][PL]
template <int NCH, int NTAPS, bool RELU, int MT>
__global__ __launch_bounds__(256) void k_conv(const _Float16* __restrict__ in,
                                              long inSamp, long inChunk,
                                              const _Float16* __restrict__ wf,
                                              const float* __restrict__ bias,
                                              _Float16* __restrict__ out, int dil) {
    const int lane = threadIdx.x & 31;
    int strip = ((int)blockIdx.x << 3) + ((int)threadIdx.x >> 5);
    const int ng = strip & 15; strip >>= 4;          // 16 L-groups of 64
    const int m  = strip % MT;
    const int b  = strip / MT;
    const int l0 = ng << 6;
    const int hi = lane >> 4;

    v8f acc[4];
    #pragma unroll
    for (int r = 0; r < 8; ++r) {
        const float bv = bias[m * 16 + hi * 8 + r];
        acc[0][r] = bv; acc[1][r] = bv; acc[2][r] = bv; acc[3][r] = bv;
    }

    const _Float16* bbase = in + (size_t)b * inSamp + (size_t)(hi * 8) * PLx + PADx + l0 + (lane & 15);
    const _Float16* wbase = wf + (size_t)lane * 16;

    #pragma unroll
    for (int ch = 0; ch < NCH; ++ch) {
        const _Float16* bch = bbase + (size_t)ch * inChunk;
        #pragma unroll
        for (int t = 0; t < NTAPS; ++t) {
            const int shift = (t - NTAPS / 2) * dil;
            const v16h a = *(const v16h*)(wbase + (size_t)((m * NCH + ch) * NTAPS + t) * FRAG);
            const _Float16* bp = bch + shift;
            v16h b0, b1, b2, b3;
            #pragma unroll
            for (int s = 0; s < 16; ++s) {
                const _Float16* q = bp + (size_t)((s < 8) ? s : (s + 8)) * PLx;
                b0[s] = q[0];
                b1[s] = q[16];
                b2[s] = q[32];
                b3[s] = q[48];
            }
            acc[0] = __builtin_amdgcn_wmma_f32_16x16x32_f16(false, a, false, b0, (short)0, acc[0], false, false);
            acc[1] = __builtin_amdgcn_wmma_f32_16x16x32_f16(false, a, false, b1, (short)0, acc[1], false, false);
            acc[2] = __builtin_amdgcn_wmma_f32_16x16x32_f16(false, a, false, b2, (short)0, acc[2], false, false);
            acc[3] = __builtin_amdgcn_wmma_f32_16x16x32_f16(false, a, false, b3, (short)0, acc[3], false, false);
        }
    }

    // C/D layout: vgpr r -> M = r + hi*8 ; N = lane&15
    _Float16* orow = out + ((size_t)b * (MT * 16) + m * 16 + hi * 8) * PLx + PADx + l0 + (lane & 15);
    #pragma unroll
    for (int t = 0; t < 4; ++t) {
        #pragma unroll
        for (int r = 0; r < 8; ++r) {
            float v = acc[t][r];
            if (RELU) v = v > 0.f ? v : 0.f;
            orow[(size_t)r * PLx + t * 16] = (_Float16)v;
        }
    }
}

// ---------------- encoder block 0 conv1 (Cin=1, K=5, dil=1) : VALU ----------------
__global__ void k_enc0(const float* __restrict__ x, const float* __restrict__ w,
                       const float* __restrict__ bias, _Float16* __restrict__ out) {
    const int idx = (int)blockIdx.x * 256 + (int)threadIdx.x;
    const int l  = idx & 1023;
    const int co = (idx >> 10) & 127;
    const int b  = idx >> 17;
    float acc = bias[co];
    #pragma unroll
    for (int k = 0; k < 5; ++k) {
        const int p = l + k - 2;
        const float xv = (p >= 0 && p < Lx) ? x[(size_t)b * 1025 + p] : 0.f;
        acc += w[co * 5 + k] * xv;
    }
    acc = acc > 0.f ? acc : 0.f;
    out[((size_t)b * 128 + co) * PLx + PADx + l] = (_Float16)acc;
}

// ---------------- avgpool(2) + nearest upsample(2) ----------------
__global__ void k_pool(const _Float16* __restrict__ enc, _Float16* __restrict__ h) {
    const int idx = (int)blockIdx.x * 256 + (int)threadIdx.x;
    const int l  = idx & 1023;
    const int ci = (idx >> 10) & 63;
    const int b  = idx >> 16;
    const size_t row = ((size_t)b * 64 + ci) * PLx + PADx;
    const int p = l & ~1;
    const float v = 0.5f * ((float)enc[row + p] + (float)enc[row + p + 1]);
    h[row + l] = (_Float16)v;
}

// ---------------- final per-sample combo 1x1 conv (192 -> 1) ----------------
__global__ void k_final(const float* __restrict__ x, const _Float16* __restrict__ ff,
                        const _Float16* __restrict__ fv, const float* __restrict__ cw,
                        const float* __restrict__ cb, float* __restrict__ outp) {
    const int idx = (int)blockIdx.x * 256 + (int)threadIdx.x;
    const int l = idx & 1023;
    const int b = idx >> 10;
    const int c = (int)x[(size_t)b * 1025 + Lx];
    const float* w = cw + (size_t)c * 192;
    float s = cb[c];
    #pragma unroll
    for (int j = 0; j < 4; ++j) {
        const _Float16* f = ff + ((size_t)(j * Bx + b)) * SB + PADx + l;
        #pragma unroll
        for (int r = 0; r < 32; ++r) s += w[j * 32 + r] * (float)f[(size_t)r * PLx];
    }
    #pragma unroll
    for (int j = 0; j < 2; ++j) {
        const _Float16* f = fv + ((size_t)((c * 2 + j) * Bx + b)) * SB + PADx + l;
        #pragma unroll
        for (int r = 0; r < 32; ++r) s += w[(4 + j) * 32 + r] * (float)f[(size_t)r * PLx];
    }
    outp[idx] = s;
}

// ---------------- host orchestration ----------------
extern "C" void kernel_launch(void* const* d_in, const int* in_sizes, int n_in,
                              void* d_out, int out_size, void* d_ws, size_t ws_size,
                              hipStream_t stream) {
    const float* x        = (const float*)d_in[0];
    const float* enc0_w1  = (const float*)d_in[1];
    const float* enc0_b1  = (const float*)d_in[2];
    const float* enc0_w2  = (const float*)d_in[3];
    const float* enc0_b2  = (const float*)d_in[4];
    const float* enc_w1   = (const float*)d_in[5];
    const float* enc_b1   = (const float*)d_in[6];
    const float* enc_w2   = (const float*)d_in[7];
    const float* enc_b2   = (const float*)d_in[8];
    const float* comp_w   = (const float*)d_in[9];
    const float* comp_b   = (const float*)d_in[10];
    const float* decf0_w1 = (const float*)d_in[11];
    const float* decf0_b1 = (const float*)d_in[12];
    const float* decf0_w2 = (const float*)d_in[13];
    const float* decf0_b2 = (const float*)d_in[14];
    const float* decf_w1  = (const float*)d_in[15];
    const float* decf_b1  = (const float*)d_in[16];
    const float* decf_w2  = (const float*)d_in[17];
    const float* decf_b2  = (const float*)d_in[18];
    const float* decv_w1  = (const float*)d_in[19];
    const float* decv_b1  = (const float*)d_in[20];
    const float* decv_w2  = (const float*)d_in[21];
    const float* decv_b2  = (const float*)d_in[22];
    const float* decv_cw  = (const float*)d_in[23];
    const float* decv_cb  = (const float*)d_in[24];

    _Float16* ws    = (_Float16*)d_ws;
    _Float16* cbuf  = ws + OFF_CBUF;
    _Float16* feate = ws + OFF_FEATE;
    _Float16* encb  = ws + OFF_ENC;
    _Float16* hdec  = ws + OFF_HDEC;
    _Float16* ff    = ws + OFF_FF;
    _Float16* fv    = ws + OFF_FV;
    _Float16* wp    = ws + OFF_WP;

    // 0) zero activation workspace (pads must be zero for SAME padding)
    k_zero<<<2048, 256, 0, stream>>>((uint4*)d_ws, (ZERO_HALVES * 2) / 16);

    // 1) pack all weights into WMMA A-fragments
    k_pack<<<25, 256, 0, stream>>>(enc_w1,   wp + FR_ENCW1  * FRAG, 5, 128, 32, 5);
    k_pack<<<1,  256, 0, stream>>>(enc0_w2,  wp + FR_ENC0W2 * FRAG, 1, 32, 128, 1);
    k_pack<<<5,  256, 0, stream>>>(enc_w2,   wp + FR_ENCW2  * FRAG, 5, 32, 128, 1);
    k_pack<<<3,  256, 0, stream>>>(comp_w,   wp + FR_COMP   * FRAG, 1, 64, 192, 1);
    k_pack<<<10, 256, 0, stream>>>(decf0_w1, wp + FR_DF0W1  * FRAG, 1, 128, 64, 5);
    k_pack<<<1,  256, 0, stream>>>(decf0_w2, wp + FR_DF0W2  * FRAG, 1, 32, 128, 1);
    k_pack<<<15, 256, 0, stream>>>(decf_w1,  wp + FR_DFW1   * FRAG, 3, 128, 32, 5);
    k_pack<<<3,  256, 0, stream>>>(decf_w2,  wp + FR_DFW2   * FRAG, 3, 32, 128, 1);
    k_pack<<<40, 256, 0, stream>>>(decv_w1,  wp + FR_DVW1   * FRAG, 8, 128, 32, 5);
    k_pack<<<8,  256, 0, stream>>>(decv_w2,  wp + FR_DVW2   * FRAG, 8, 32, 128, 1);

    const long S128 = 128L * PLx, S64 = 64L * PLx, S32 = 32L * PLx;

    // 2) encoder block 0
    k_enc0<<<32768, 256, 0, stream>>>(x, enc0_w1, enc0_b1, cbuf);
    k_conv<4, 1, true, 2><<<256, 256, 0, stream>>>(cbuf, S128, S32, wp + FR_ENC0W2 * FRAG,
                                                   enc0_b2, feate, 1);
    // encoder blocks 1..5 (dil = 2^j)
    for (int j = 1; j < 6; ++j) {
        k_conv<1, 5, true, 8><<<1024, 256, 0, stream>>>(feate + (long)(j - 1) * Bx * SB, SB, SB,
            wp + (FR_ENCW1 + (j - 1) * 40) * FRAG, enc_b1 + (j - 1) * 128, cbuf, 1 << j);
        k_conv<4, 1, true, 2><<<256, 256, 0, stream>>>(cbuf, S128, S32,
            wp + (FR_ENCW2 + (j - 1) * 8) * FRAG, enc_b2 + (j - 1) * 32,
            feate + (long)j * Bx * SB, 1);
    }

    // 3) comp 1x1 over concat(6 x 32ch), then avgpool+upsample
    k_conv<6, 1, false, 4><<<512, 256, 0, stream>>>(feate, SB, (long)Bx * SB,
        wp + FR_COMP * FRAG, comp_b, encb, 1);
    k_pool<<<16384, 256, 0, stream>>>(encb, hdec);

    // 4) fixed decoder block 0 (Cin=64, dil=32)
    k_conv<2, 5, true, 8><<<1024, 256, 0, stream>>>(hdec, S64, S32,
        wp + FR_DF0W1 * FRAG, decf0_b1, cbuf, 32);
    k_conv<4, 1, true, 2><<<256, 256, 0, stream>>>(cbuf, S128, S32,
        wp + FR_DF0W2 * FRAG, decf0_b2, ff, 1);
    // fixed decoder blocks 1..3 (dil = 16, 8, 4)
    for (int j = 1; j < 4; ++j) {
        k_conv<1, 5, true, 8><<<1024, 256, 0, stream>>>(ff + (long)(j - 1) * Bx * SB, SB, SB,
            wp + (FR_DFW1 + (j - 1) * 40) * FRAG, decf_b1 + (j - 1) * 128, cbuf, 32 >> j);
        k_conv<4, 1, true, 2><<<256, 256, 0, stream>>>(cbuf, S128, S32,
            wp + (FR_DFW2 + (j - 1) * 8) * FRAG, decf_b2 + (j - 1) * 32,
            ff + (long)j * Bx * SB, 1);
    }

    // 5) variable decoder: dense over 4 combos, 2 blocks each (dil = 2, 1)
    for (int c = 0; c < 4; ++c) {
        for (int jj = 0; jj < 2; ++jj) {
            const int layer = c * 2 + jj;
            const _Float16* inb = (jj == 0) ? (ff + 3L * Bx * SB)
                                            : (fv + (long)(c * 2) * Bx * SB);
            k_conv<1, 5, true, 8><<<1024, 256, 0, stream>>>(inb, SB, SB,
                wp + (FR_DVW1 + layer * 40) * FRAG, decv_b1 + layer * 128, cbuf, 2 >> jj);
            k_conv<4, 1, true, 2><<<256, 256, 0, stream>>>(cbuf, S128, S32,
                wp + (FR_DVW2 + layer * 8) * FRAG, decv_b2 + layer * 32,
                fv + (long)layer * Bx * SB, 1);
        }
    }

    // 6) per-sample combo select + 1x1 (192 -> 1)
    k_final<<<256, 256, 0, stream>>>(x, ff, fv, decv_cw, decv_cb, (float*)d_out);
}
